// AR_RNN_LSTM_22703197127170
// MI455X (gfx1250) — matrix operand order repaired and checked by reading, not compiled, not run on hardware
//
#include <hip/hip_runtime.h>

typedef __attribute__((ext_vector_type(16))) _Float16 v16h;
typedef __attribute__((ext_vector_type(8)))  _Float16 v8h;
typedef __attribute__((ext_vector_type(8)))  float    v8f;

namespace {
constexpr int kUnits   = 512;
constexpr int kDdim    = 256;
constexpr int kSteps   = 100;          // IN_STEPS == OUT_STEPS == 100
constexpr int kBatch   = 1024;
constexpr int kThreads = 512;          // 16 waves
constexpr int kTPW     = 2;            // h-tiles per wave (16 waves * 2 = 32 tiles)
constexpr int kLdsK    = 520;          // padded f16 row stride for 512-wide h buffers
constexpr int kSxK     = 264;          // padded f16 row stride for 256-wide x/pred buffer

// packed f16 weight offsets (elements) inside d_ws
constexpr size_t oW0 = 0;
constexpr size_t oU0 = oW0 + (size_t)256 * 2048;
constexpr size_t oW1 = oU0 + (size_t)512 * 2048;
constexpr size_t oU1 = oW1 + (size_t)512 * 2048;
constexpr size_t oW2 = oU1 + (size_t)512 * 2048;
constexpr size_t oU2 = oW2 + (size_t)512 * 2048;
constexpr size_t oWd = oU2 + (size_t)512 * 2048;
}

// ---------------------------------------------------------------------------
// Repack one f32 weight matrix [K, N] (row-major) into the WMMA B-operand
// layout, f16:  packed[((nt*KB + kb)*32 + lane)*16 + e] = W[k, n]
//   k = kb*32 + 16*(lane>>4) + e ,  n = nt*16 + (lane&15)
// so each lane's 16 halves (32B) for one 32x16 K-tile are contiguous.
// ---------------------------------------------------------------------------
__global__ void wpack_kernel(const float* __restrict__ src,
                             _Float16* __restrict__ dst, int K, int N) {
    int p = blockIdx.x * 256 + threadIdx.x;
    if (p >= K * N) return;
    int e    = p & 15;
    int lane = (p >> 4) & 31;
    int idx  = p >> 9;
    int KB   = K >> 5;
    int kb   = idx % KB;
    int nt   = idx / KB;
    int k    = kb * 32 + ((lane >> 4) << 4) + e;
    int n    = nt * 16 + (lane & 15);
    dst[p] = (_Float16)src[(size_t)k * N + n];
}

// ---------------------------------------------------------------------------
// helpers
// ---------------------------------------------------------------------------
__device__ __forceinline__ v8f wmma_f16(v16h a, v16h b, v8f c) {
    return __builtin_amdgcn_wmma_f32_16x16x32_f16(false, a, false, b,
                                                  (short)0, c, false, false);
}

// A operand (16x32 f16) for this lane from LDS row-major activations.
// row = lane&15 (caller passes that row's base pointer).
// Lane needs K = kb*32 + 16*(lane>>4) + {0..15}  -> two b128 reads.
__device__ __forceinline__ v16h loadA(const _Float16* row, int kb, int lh) {
    int k0 = kb * 32 + lh * 16;
    v8h lo = *(const v8h*)(row + k0);
    v8h hi = *(const v8h*)(row + k0 + 8);
    return __builtin_shufflevector(lo, hi, 0, 1, 2, 3, 4, 5, 6, 7,
                                   8, 9, 10, 11, 12, 13, 14, 15);
}

// B operand (32x16 f16) from packed weights: one 32B load per lane.
__device__ __forceinline__ v16h loadB(const _Float16* Wp, int nt, int kb,
                                      int KB, int lane) {
    return *(const v16h*)(Wp + ((((size_t)nt * KB + kb) * 32 + lane) << 4));
}

__device__ __forceinline__ float sigm_f(float x) {
    return 1.0f / (1.0f + __expf(-x));
}
__device__ __forceinline__ float tanh_fast(float x) {
    return 1.0f - 2.0f / (__expf(2.0f * x) + 1.0f);
}

// ---------------------------------------------------------------------------
// One LSTM layer step for a 16-row batch tile.
// Each wave owns kTPW h-column tiles (tw = wave*kTPW+j); gate g's z-tile index
// is g*32+tw in the 2048-wide pre-activation.  c state is register-resident in
// WMMA D-layout.  Ends with the new h written to hBuf (f16, row-major).
// ---------------------------------------------------------------------------
template <int KXB>
__device__ __forceinline__ void lstm_layer(const _Float16* aInRow,   // &act[ln][0]
                                           const _Float16* hInRow,   // &hBuf[ln][0]
                                           _Float16* hBase,          // &hBuf[0][0]
                                           const _Float16* __restrict__ Wp,
                                           const _Float16* __restrict__ Up,
                                           const float* __restrict__ bias,
                                           v8f* cst, int wave, int lane) {
    const int ln = lane & 15;
    const int lh = lane >> 4;
    v8f hOut[kTPW];
#pragma unroll
    for (int j = 0; j < kTPW; ++j) {
        const int tw = wave * kTPW + j;
        v8f zi = {0, 0, 0, 0, 0, 0, 0, 0};
        v8f zf = {0, 0, 0, 0, 0, 0, 0, 0};
        v8f zg = {0, 0, 0, 0, 0, 0, 0, 0};
        v8f zo = {0, 0, 0, 0, 0, 0, 0, 0};
#pragma unroll 2
        for (int kb = 0; kb < KXB; ++kb) {          // feed-forward part
            v16h a = loadA(aInRow, kb, lh);
            zi = wmma_f16(a, loadB(Wp, 0 * 32 + tw, kb, KXB, lane), zi);
            zf = wmma_f16(a, loadB(Wp, 1 * 32 + tw, kb, KXB, lane), zf);
            zg = wmma_f16(a, loadB(Wp, 2 * 32 + tw, kb, KXB, lane), zg);
            zo = wmma_f16(a, loadB(Wp, 3 * 32 + tw, kb, KXB, lane), zo);
        }
#pragma unroll 2
        for (int kb = 0; kb < 16; ++kb) {           // recurrent part, K=512
            v16h a = loadA(hInRow, kb, lh);
            zi = wmma_f16(a, loadB(Up, 0 * 32 + tw, kb, 16, lane), zi);
            zf = wmma_f16(a, loadB(Up, 1 * 32 + tw, kb, 16, lane), zf);
            zg = wmma_f16(a, loadB(Up, 2 * 32 + tw, kb, 16, lane), zg);
            zo = wmma_f16(a, loadB(Up, 3 * 32 + tw, kb, 16, lane), zo);
        }
        const int col = tw * 16 + ln;
        float bi = bias[0 * 512 + col];
        float bf = bias[1 * 512 + col];
        float bg = bias[2 * 512 + col];
        float bo = bias[3 * 512 + col];
#pragma unroll
        for (int r = 0; r < 8; ++r) {
            float iv = sigm_f(zi[r] + bi);
            float fv = sigm_f(zf[r] + bf);
            float gv = tanh_fast(zg[r] + bg);
            float ov = sigm_f(zo[r] + bo);
            float cc = fv * cst[j][r] + iv * gv;
            cst[j][r]  = cc;
            hOut[j][r] = ov * tanh_fast(cc);
        }
    }
    __syncthreads();   // all waves done reading act/h before h is overwritten
#pragma unroll
    for (int j = 0; j < kTPW; ++j) {
        const int tw = wave * kTPW + j;
#pragma unroll
        for (int r = 0; r < 8; ++r)
            hBase[(size_t)(8 * lh + r) * kLdsK + tw * 16 + ln] =
                (_Float16)hOut[j][r];
    }
    __syncthreads();
}

// Dense head: pred[16,256] = h2 @ Wd + bd.  One n-tile per wave (nt = wave).
__device__ __forceinline__ void dense_head(const _Float16* hInRow,
                                           const _Float16* __restrict__ Wdp,
                                           const float* __restrict__ bd,
                                           int wave, int lane, v8f& o0) {
    const int ln = lane & 15;
    const int lh = lane >> 4;
    v8f a0 = {0, 0, 0, 0, 0, 0, 0, 0};
#pragma unroll 4
    for (int kb = 0; kb < 16; ++kb) {
        v16h a = loadA(hInRow, kb, lh);
        a0 = wmma_f16(a, loadB(Wdp, wave, kb, 16, lane), a0);
    }
    float b0 = bd[wave * 16 + ln];
#pragma unroll
    for (int r = 0; r < 8; ++r) o0[r] = a0[r] + b0;
}

// ---------------------------------------------------------------------------
// Persistent per-batch-tile kernel: 64 WGs x 16 rows, full 200-step rollout.
// 16 waves per WG -> 4 waves per SIMD for latency hiding; c state stays in
// registers (48 VGPRs/wave) so the wave fits in <=256 architectural VGPRs.
// ---------------------------------------------------------------------------
__launch_bounds__(kThreads, 1)
__global__ void lstm_ar_kernel(const float* __restrict__ x,
                               const float* __restrict__ bias0,
                               const float* __restrict__ bias1,
                               const float* __restrict__ bias2,
                               const float* __restrict__ bd,
                               const _Float16* __restrict__ wp,
                               float* __restrict__ out) {
    const int tid  = threadIdx.x;
    const int wave = tid >> 5;
    const int lane = tid & 31;
    const int ln   = lane & 15;
    const int lh   = lane >> 4;
    const int rb   = blockIdx.x * 16;

    __shared__ __attribute__((aligned(32))) _Float16 sx [16][kSxK];
    __shared__ __attribute__((aligned(32))) _Float16 sh0[16][kLdsK];
    __shared__ __attribute__((aligned(32))) _Float16 sh1[16][kLdsK];
    __shared__ __attribute__((aligned(32))) _Float16 sh2[16][kLdsK];

    for (int i = tid; i < 16 * kLdsK; i += kThreads) {
        (&sh0[0][0])[i] = (_Float16)0.0f;
        (&sh1[0][0])[i] = (_Float16)0.0f;
        (&sh2[0][0])[i] = (_Float16)0.0f;
    }
    __syncthreads();

    const v8f z8 = {0, 0, 0, 0, 0, 0, 0, 0};
    v8f c0[kTPW] = {z8, z8};
    v8f c1[kTPW] = {z8, z8};
    v8f c2[kTPW] = {z8, z8};
    v8f p0 = z8;                       // pred0 stash (registers, D-layout)

    const _Float16* W0p = wp + oW0;
    const _Float16* U0p = wp + oU0;
    const _Float16* W1p = wp + oW1;
    const _Float16* U1p = wp + oU1;
    const _Float16* W2p = wp + oW2;
    const _Float16* U2p = wp + oU2;
    const _Float16* Wdp = wp + oWd;

    const _Float16* axRow = &sx[ln][0];
    const _Float16* h0Row = &sh0[ln][0];
    const _Float16* h1Row = &sh1[ln][0];
    const _Float16* h2Row = &sh2[ln][0];

    // -------------------------- warmup: t = 0..99 --------------------------
    for (int t = 0; t < kSteps; ++t) {
#pragma unroll
        for (int i = 0; i < 8; ++i) {                // stage x[t] -> LDS f16
            int idx = tid + i * kThreads;
            int row = idx >> 8;
            int k   = idx & 255;
            sx[row][k] = (_Float16)x[(size_t)(rb + row) * kSteps * kDdim +
                                     (size_t)t * kDdim + k];
        }
        __syncthreads();
        lstm_layer<8> (axRow, h0Row, &sh0[0][0], W0p, U0p, bias0, c0, wave, lane);
        lstm_layer<16>(h0Row, h1Row, &sh1[0][0], W1p, U1p, bias1, c1, wave, lane);
        lstm_layer<16>(h1Row, h2Row, &sh2[0][0], W2p, U2p, bias2, c2, wave, lane);
        if (t == 0) {                                // pred0 from h2 at t=0
            dense_head(h2Row, Wdp, bd, wave, lane, p0);
#pragma unroll
            for (int r = 0; r < 8; ++r) {
                size_t ro = (size_t)(rb + 8 * lh + r) * kSteps * kDdim;
                out[ro + wave * 16 + ln] = p0[r];
            }
        }
    }

    // seed AR input with pred0
#pragma unroll
    for (int r = 0; r < 8; ++r)
        sx[8 * lh + r][wave * 16 + ln] = (_Float16)p0[r];
    __syncthreads();

    // ------------------------ autoregressive: s = 1..99 --------------------
    for (int s = 1; s < kSteps; ++s) {
        lstm_layer<8> (axRow, h0Row, &sh0[0][0], W0p, U0p, bias0, c0, wave, lane);
        lstm_layer<16>(h0Row, h1Row, &sh1[0][0], W1p, U1p, bias1, c1, wave, lane);
        lstm_layer<16>(h1Row, h2Row, &sh2[0][0], W2p, U2p, bias2, c2, wave, lane);
        v8f q0;
        dense_head(h2Row, Wdp, bd, wave, lane, q0);
#pragma unroll
        for (int r = 0; r < 8; ++r) {
            size_t ro = (size_t)(rb + 8 * lh + r) * kSteps * kDdim +
                        (size_t)s * kDdim;
            out[ro + wave * 16 + ln] = q0[r];
            sx[8 * lh + r][wave * 16 + ln] = (_Float16)q0[r];
        }
        __syncthreads();
    }
}

// ---------------------------------------------------------------------------
extern "C" void kernel_launch(void* const* d_in, const int* in_sizes, int n_in,
                              void* d_out, int out_size, void* d_ws,
                              size_t ws_size, hipStream_t stream) {
    const float* x  = (const float*)d_in[0];
    const float* W0 = (const float*)d_in[1];
    const float* U0 = (const float*)d_in[2];
    const float* b0 = (const float*)d_in[3];
    const float* W1 = (const float*)d_in[4];
    const float* U1 = (const float*)d_in[5];
    const float* b1 = (const float*)d_in[6];
    const float* W2 = (const float*)d_in[7];
    const float* U2 = (const float*)d_in[8];
    const float* b2 = (const float*)d_in[9];
    const float* Wd = (const float*)d_in[10];
    const float* bd = (const float*)d_in[11];
    float* out = (float*)d_out;
    _Float16* ws = (_Float16*)d_ws;

    auto pack = [&](const float* src, size_t off, int K, int N) {
        int n = K * N;
        wpack_kernel<<<(n + 255) / 256, 256, 0, stream>>>(src, ws + off, K, N);
    };
    pack(W0, oW0, 256, 2048);
    pack(U0, oU0, 512, 2048);
    pack(W1, oW1, 512, 2048);
    pack(U1, oU1, 512, 2048);
    pack(W2, oW2, 512, 2048);
    pack(U2, oU2, 512, 2048);
    pack(Wd, oWd, 512, 256);

    lstm_ar_kernel<<<kBatch / 16, kThreads, 0, stream>>>(x, b0, b1, b2, bd, ws,
                                                         out);
}
